// ContrastiveLearningModel_87153476370492
// MI455X (gfx1250) — matrix operand order repaired and checked by reading, not compiled, not run on hardware
//
#include <hip/hip_runtime.h>

// ---------------------------------------------------------------------------
// Types for CDNA5 WMMA (wave32): A/B = 16 bf16 per lane (8 VGPRs), C/D = 8 f32
// ---------------------------------------------------------------------------
typedef __attribute__((ext_vector_type(8)))  float        v8f;
typedef __attribute__((ext_vector_type(16))) __bf16       v16bf;
typedef __attribute__((ext_vector_type(4)))  unsigned int v4u;
struct U32x8 { v4u a, b; };   // 32 bytes == 16 bf16 == one WMMA operand fragment

static __device__ __forceinline__ unsigned short f2bf(float f) {
  // round-to-nearest-even f32 -> bf16 (inputs finite)
  unsigned int u = __float_as_uint(f);
  u += 0x7fffu + ((u >> 16) & 1u);
  return (unsigned short)(u >> 16);
}

static __device__ __forceinline__ v16bf frag_from(const unsigned short* p0,
                                                  const unsigned short* p1) {
  U32x8 u;
  u.a = *reinterpret_cast<const v4u*>(p0);   // 8 bf16
  u.b = *reinterpret_cast<const v4u*>(p1);   // 8 bf16
  return __builtin_bit_cast(v16bf, u);
}

// ---------------------------------------------------------------------------
// One-shot f32 -> bf16 weight conversion (optionally K-padded with zeros)
// ---------------------------------------------------------------------------
__global__ void cvt_bf16_kernel(const float* __restrict__ src,
                                unsigned short* __restrict__ dst, int n) {
  int i = blockIdx.x * blockDim.x + threadIdx.x;
  if (i < n) dst[i] = f2bf(src[i]);
}

__global__ void cvt_bf16_pad_kernel(const float* __restrict__ src,
                                    unsigned short* __restrict__ dst,
                                    int rows, int cin, int cpad) {
  int i = blockIdx.x * blockDim.x + threadIdx.x;
  if (i >= rows * cpad) return;
  int r = i / cpad, c = i % cpad;
  dst[i] = (c < cin) ? f2bf(src[(size_t)r * cin + c]) : (unsigned short)0;
}

// ---------------------------------------------------------------------------
// Furthest point sampling: one block per batch-frame. Matches jnp scan:
// out[j] = last; mind = min(mind, d2(last)); last = argmax(mind) (first idx tie)
// Writes anchor coords directly into the (padded) next-layer xyz buffer.
// ---------------------------------------------------------------------------
#define FPS_MAXN 2048
__global__ void fps_kernel(const float* __restrict__ xyz, int Lpad, int t, int N, int m,
                           float* __restrict__ xyzNext, int Lnext, int fout) {
  __shared__ float sx[FPS_MAXN], sy[FPS_MAXN], sz[FPS_MAXN], dmin[FPS_MAXN];
  __shared__ float rv[256];
  __shared__ int   ri[256];
  int bf = blockIdx.x, tid = threadIdx.x;
  const float* p = xyz + (((size_t)bf * Lpad + t) * (size_t)N) * 3;
  float* anch = xyzNext + (((size_t)bf * Lnext + fout) * (size_t)m) * 3;
  for (int j = tid; j < N; j += 256) {
    sx[j] = p[3*j]; sy[j] = p[3*j+1]; sz[j] = p[3*j+2]; dmin[j] = 1e10f;
  }
  __syncthreads();
  int last = 0;
  for (int it = 0; it < m; ++it) {
    if (tid == 0) { anch[3*it] = sx[last]; anch[3*it+1] = sy[last]; anch[3*it+2] = sz[last]; }
    float lx = sx[last], ly = sy[last], lz = sz[last];
    float bv = -1.0f; int bi = 0;
    for (int j = tid; j < N; j += 256) {
      float dx = sx[j]-lx, dy = sy[j]-ly, dz = sz[j]-lz;
      float d2 = dx*dx + dy*dy + dz*dz;
      float dm = dmin[j]; dm = d2 < dm ? d2 : dm; dmin[j] = dm;
      if (dm > bv) { bv = dm; bi = j; }   // strict > keeps lowest j per thread
    }
    rv[tid] = bv; ri[tid] = bi;
    __syncthreads();
    for (int s = 128; s > 0; s >>= 1) {
      if (tid < s) {
        float ov = rv[tid+s]; int oi = ri[tid+s];
        if (ov > rv[tid] || (ov == rv[tid] && oi < ri[tid])) { rv[tid] = ov; ri[tid] = oi; }
      }
      __syncthreads();
    }
    last = ri[0];
    __syncthreads();
  }
}

// ---------------------------------------------------------------------------
// Ball query, pointnet2 semantics: first K(=9) indices with d2 < r2; remaining
// slots filled with the first in-radius index (0 if none). K padded to 16 with
// idx=0 / disp=0 so the WMMA M-tile is full (pad rows die via d=0 in epilogue).
// ---------------------------------------------------------------------------
__global__ void ballquery_kernel(const float* __restrict__ xyz, int Lpad, int frame, int N,
                                 const float* __restrict__ xyzNext, int Lnext, int fout, int m,
                                 float r2, int* __restrict__ idx16, float* __restrict__ disp16,
                                 int rows) {
  int gid = blockIdx.x * blockDim.x + threadIdx.x;
  if (gid >= rows) return;
  int bf = gid / m, a = gid % m;
  const float* p  = xyz + (((size_t)bf * Lpad + frame) * (size_t)N) * 3;
  const float* an = xyzNext + (((size_t)bf * Lnext + fout) * (size_t)m + a) * 3;
  float ax = an[0], ay = an[1], az = an[2];
  int idxs[9]; int cnt = 0;
  for (int j = 0; j < N && cnt < 9; ++j) {
    float dx = p[3*j]-ax, dy = p[3*j+1]-ay, dz = p[3*j+2]-az;
    if (dx*dx + dy*dy + dz*dz < r2) idxs[cnt++] = j;
  }
  int first = (cnt > 0) ? idxs[0] : 0;
  for (int k = cnt; k < 9; ++k) idxs[k] = first;
  int*   oi = idx16  + (size_t)gid * 16;
  float* od = disp16 + (size_t)gid * 48;
  for (int k = 0; k < 9; ++k) {
    int j = idxs[k]; oi[k] = j;
    od[3*k] = p[3*j]-ax; od[3*k+1] = p[3*j+1]-ay; od[3*k+2] = p[3*j+2]-az;
  }
  for (int k = 9; k < 16; ++k) { oi[k] = 0; od[3*k] = 0.f; od[3*k+1] = 0.f; od[3*k+2] = 0.f; }
}

// ---------------------------------------------------------------------------
// Layer 0 (no input features): part[a,o] = sum_k Wd[o,:] . disp[a,k,:]
// ---------------------------------------------------------------------------
__global__ void disp_part_kernel(const float* __restrict__ disp16, const float* __restrict__ Wd,
                                 float* __restrict__ nf, int mid, int Ctot, int coff, int rows) {
  int gid = blockIdx.x * blockDim.x + threadIdx.x;
  if (gid >= rows * mid) return;
  int ag = gid / mid, o = gid % mid;
  const float* dp = disp16 + (size_t)ag * 48;
  float w0 = Wd[3*o], w1 = Wd[3*o+1], w2 = Wd[3*o+2];
  float s = 0.f;
  for (int k = 0; k < 9; ++k)
    s += w0*dp[3*k] + w1*dp[3*k+1] + w2*dp[3*k+2];
  nf[(size_t)ag * Ctot + coff + o] = s;
}

// ---------------------------------------------------------------------------
// Fused gather + (fg @ Wf^T) * (Wd @ disp) + sum_k via v_wmma_f32_16x16x32_bf16.
// One wave per (anchor, 32-wide mid group): the gathered A fragment (the
// expensive scattered loads) is reused by TWO B fragments / accumulators.
// A rows = 16 padded neighbors (M=k); pad rows are annihilated by d=0.
// Epilogue: C[k,o] *= Wd.disp[k], reduce over k (8 acc rows + shfl_xor 16).
// ---------------------------------------------------------------------------
__global__ void feat_gemm_kernel(const unsigned short* __restrict__ feats, int Lpad, int frame,
                                 int N, int Cin,
                                 const int* __restrict__ idx16, const float* __restrict__ disp16,
                                 const unsigned short* __restrict__ Wf, const float* __restrict__ Wd,
                                 int mid, float* __restrict__ nf, int Ctot, int coff,
                                 int m, int ntiles) {
  int wid = (blockIdx.x * blockDim.x + threadIdx.x) >> 5;
  if (wid >= ntiles) return;                       // wave-uniform: EXEC stays all-1s
  int ogroups = mid >> 5;                          // 32 output channels per wave
  int on = wid % ogroups;
  int ag = wid / ogroups;                          // ag = bf*m + a
  int bf = ag / m;
  int l  = threadIdx.x & 31;
  int lm = l & 15, hi = l >> 4;
  int ridx = idx16[(size_t)ag * 16 + lm];          // neighbor row for A (M = lm = k)
  const unsigned short* frow =
      feats + (((size_t)bf * Lpad + frame) * (size_t)N + ridx) * (size_t)Cin;
  int o0 = on * 32 + lm;                           // B columns (N = lm) for tile 0/1
  int o1 = o0 + 16;
  const unsigned short* wrow0 = Wf + (size_t)o0 * Cin;
  const unsigned short* wrow1 = Wf + (size_t)o1 * Cin;
  int kbA = hi * 16, kbB = hi * 16;
  v8f acc0 = {0.f,0.f,0.f,0.f,0.f,0.f,0.f,0.f};
  v8f acc1 = {0.f,0.f,0.f,0.f,0.f,0.f,0.f,0.f};
  for (int c0 = 0; c0 < Cin; c0 += 32) {
    v16bf av  = frag_from(frow  + c0 + kbA, frow  + c0 + kbA + 8);
    v16bf bv0 = frag_from(wrow0 + c0 + kbB, wrow0 + c0 + kbB + 8);
    v16bf bv1 = frag_from(wrow1 + c0 + kbB, wrow1 + c0 + kbB + 8);
    acc0 = __builtin_amdgcn_wmma_f32_16x16x32_bf16(false, av, false, bv0, (short)0, acc0, false, false);
    acc1 = __builtin_amdgcn_wmma_f32_16x16x32_bf16(false, av, false, bv1, (short)0, acc1, false, false);
  }
  // epilogue: multiply by d[k,o] = Wd[o,:].disp[k,:] and reduce over k
  float w00 = Wd[3*o0], w01 = Wd[3*o0+1], w02 = Wd[3*o0+2];
  float w10 = Wd[3*o1], w11 = Wd[3*o1+1], w12 = Wd[3*o1+2];
  const float* db = disp16 + (size_t)ag * 48;
  float s0 = 0.f, s1 = 0.f;
#pragma unroll
  for (int r = 0; r < 8; ++r) {
    int k = r + hi * 8;                            // C row M = r + half*8
    float d0 = db[3*k], d1 = db[3*k+1], d2 = db[3*k+2];
    s0 += acc0[r] * (w00*d0 + w01*d1 + w02*d2);
    s1 += acc1[r] * (w10*d0 + w11*d1 + w12*d2);
  }
  s0 += __shfl_xor(s0, 16, 32);
  s1 += __shfl_xor(s1, 16, 32);
  if (hi == 0) {
    float* outp = nf + (size_t)ag * Ctot + coff;
    outp[o0] = s0;
    outp[o1] = s1;
  }
}

// ---------------------------------------------------------------------------
// BatchNorm (batch stats, biased var) + ReLU. Reads f32 nf, writes bf16 copy
// (row stride Cpad) consumed by the temporal WMMA GEMM. One block per channel.
// ---------------------------------------------------------------------------
__global__ void bn_relu_kernel(const float* __restrict__ nf, unsigned short* __restrict__ nfb,
                               int rows, int C, int Cpad) {
  __shared__ float s1[256], s2[256];
  int c = blockIdx.x, tid = threadIdx.x;
  float sum = 0.f, sq = 0.f;
  for (int r = tid; r < rows; r += 256) {
    float v = nf[(size_t)r * C + c]; sum += v; sq += v * v;
  }
  s1[tid] = sum; s2[tid] = sq;
  __syncthreads();
  for (int s = 128; s > 0; s >>= 1) {
    if (tid < s) { s1[tid] += s1[tid+s]; s2[tid] += s2[tid+s]; }
    __syncthreads();
  }
  float mu  = s1[0] / (float)rows;
  float var = s2[0] / (float)rows - mu * mu;
  float inv = rsqrtf(var + 1e-5f);
  for (int r = tid; r < rows; r += 256) {
    float v = (nf[(size_t)r * C + c] - mu) * inv;
    v = v > 0.f ? v : 0.f;
    nfb[(size_t)r * Cpad + c] = f2bf(v);
  }
}

// ---------------------------------------------------------------------------
// Generic GEMM: out[r,o] = act( sum_c A[r,c]*W[o,c] + bias[o] ). A and W are
// bf16 with K padded to a multiple of 32 -> no guards, pure b128 loads + WMMA.
// 1x2 register blocking: one A fragment feeds two B fragments / accumulators.
// Writes either f32 (final output) or bf16 (inter-layer feats).
// Row r maps to [((r/m)*Lnext + fout)*m + r%m]*Cout (padded frame layout).
// ---------------------------------------------------------------------------
__global__ void gemm_wmma_kernel(const unsigned short* __restrict__ A, int lda,
                                 const unsigned short* __restrict__ W, int Cin, int Cout,
                                 const float* __restrict__ bias, int dorelu,
                                 float* __restrict__ outF32, unsigned short* __restrict__ outB16,
                                 int m, int Lnext, int fout, int ntiles, int tM) {
  int wid = (blockIdx.x * blockDim.x + threadIdx.x) >> 5;
  if (wid >= ntiles) return;
  int tm = wid % tM, tn = wid / tM;                // tn: 32-column group
  int l  = threadIdx.x & 31;
  int lm = l & 15, hi = l >> 4;
  int r  = tm * 16 + lm;
  const unsigned short* arow = A + (size_t)r * lda;
  int oc0 = tn * 32 + lm;
  int oc1 = oc0 + 16;
  const unsigned short* wrow0 = W + (size_t)oc0 * Cin;
  const unsigned short* wrow1 = W + (size_t)oc1 * Cin;
  int kbA = hi * 16, kbB = hi * 16;
  v8f acc0 = {0.f,0.f,0.f,0.f,0.f,0.f,0.f,0.f};
  v8f acc1 = {0.f,0.f,0.f,0.f,0.f,0.f,0.f,0.f};
  for (int c0 = 0; c0 < Cin; c0 += 32) {
    v16bf av  = frag_from(arow  + c0 + kbA, arow  + c0 + kbA + 8);
    v16bf bv0 = frag_from(wrow0 + c0 + kbB, wrow0 + c0 + kbB + 8);
    v16bf bv1 = frag_from(wrow1 + c0 + kbB, wrow1 + c0 + kbB + 8);
    acc0 = __builtin_amdgcn_wmma_f32_16x16x32_bf16(false, av, false, bv0, (short)0, acc0, false, false);
    acc1 = __builtin_amdgcn_wmma_f32_16x16x32_bf16(false, av, false, bv1, (short)0, acc1, false, false);
  }
  float b0 = bias ? bias[oc0] : 0.f;
  float b1 = bias ? bias[oc1] : 0.f;
#pragma unroll
  for (int vr = 0; vr < 8; ++vr) {
    int rr = tm * 16 + vr + hi * 8;                // C row M = vr + half*8
    float v0 = acc0[vr] + b0;
    float v1 = acc1[vr] + b1;
    if (dorelu) { v0 = v0 > 0.f ? v0 : 0.f; v1 = v1 > 0.f ? v1 : 0.f; }
    int bfb = rr / m, a = rr % m;
    size_t obase = (((size_t)bfb * Lnext + fout) * (size_t)m + a) * (size_t)Cout;
    if (outF32) { outF32[obase + oc0] = v0; outF32[obase + oc1] = v1; }
    else        { outB16[obase + oc0] = f2bf(v0); outB16[obase + oc1] = f2bf(v1); }
  }
}

// ---------------------------------------------------------------------------
// Host-side schedule
// ---------------------------------------------------------------------------
struct Cfg { int cin, mid, cout; float r; int ss, tk, ts, p0, p1; };
static const Cfg CFGS[6] = {
  {   0,   45,   64, 1.5f, 2, 1, 1, 0, 0},
  {  64,   96,  128, 3.0f, 2, 3, 2, 1, 0},
  { 128,  192,  256, 3.0f, 1, 3, 1, 1, 1},
  { 256,  384,  512, 6.0f, 2, 3, 2, 1, 0},
  { 512,  768, 1024, 6.0f, 1, 3, 1, 1, 1},
  {1024, 1536, 2048, 6.0f, 2, 1, 1, 0, 0},
};

extern "C" void kernel_launch(void* const* d_in, const int* in_sizes, int n_in,
                              void* d_out, int out_size, void* d_ws, size_t ws_size,
                              hipStream_t stream) {
  (void)in_sizes; (void)n_in; (void)out_size; (void)ws_size;
  // setup_inputs() dict order: xyzs, epoch, then per layer d, [f], t, then fc_w, fc_b
  const float* xyzs  = (const float*)d_in[0];
  const float* Wd[6] = {(const float*)d_in[2], (const float*)d_in[4], (const float*)d_in[7],
                        (const float*)d_in[10], (const float*)d_in[13], (const float*)d_in[16]};
  const float* Wf[6] = {nullptr, (const float*)d_in[5], (const float*)d_in[8],
                        (const float*)d_in[11], (const float*)d_in[14], (const float*)d_in[17]};
  const float* Wt[6] = {(const float*)d_in[3], (const float*)d_in[6], (const float*)d_in[9],
                        (const float*)d_in[12], (const float*)d_in[15], (const float*)d_in[18]};
  const float* fc_w = (const float*)d_in[19];
  const float* fc_b = (const float*)d_in[20];

  // Workspace carving (deterministic offsets; ~200 MB total)
  char* wp = (char*)d_ws;
  auto carve = [&](size_t nbytes) -> void* {
    void* r = (void*)wp; wp += (nbytes + 255) & ~(size_t)255; return r;
  };
  float*          xyzA  = (float*)carve(  524288ull * 4); // max padded xyz: 32*5*1024*3
  float*          xyzB  = (float*)carve(  524288ull * 4);
  unsigned short* featA = (unsigned short*)carve(12582912ull * 2); // bf16 feats ping
  unsigned short* featB = (unsigned short*)carve(12582912ull * 2); // bf16 feats pong
  float*          nf    = (float*)carve(18874368ull * 4); // max nf: 8192 x 2304 (f32)
  unsigned short* nfb   = (unsigned short*)carve(18874368ull * 2); // bf16 copy for WMMA
  int*            idx16 = (int*)  carve(  524288ull * 4); // 32*1024*16
  float*          disp16= (float*)carve( 1572864ull * 4); // 32*1024*16*3
  // bf16 weight arenas
  unsigned short* wfB[6] = {nullptr,
    (unsigned short*)carve(   6144ull * 2), (unsigned short*)carve(  24576ull * 2),
    (unsigned short*)carve(  98304ull * 2), (unsigned short*)carve( 393216ull * 2),
    (unsigned short*)carve(1572864ull * 2)};
  unsigned short* wtB[6] = {
    (unsigned short*)carve(   4096ull * 2), // Wt0 padded 64x64
    (unsigned short*)carve(  36864ull * 2), (unsigned short*)carve( 147456ull * 2),
    (unsigned short*)carve( 589824ull * 2), (unsigned short*)carve(2359296ull * 2),
    (unsigned short*)carve(3145728ull * 2)};
  unsigned short* fcwB = (unsigned short*)carve(2097152ull * 2);

  // --- one-shot weight conversions to bf16 -------------------------------
  for (int li = 1; li < 6; ++li) {
    int n = CFGS[li].mid * CFGS[li].cin;
    cvt_bf16_kernel<<<(n + 255) / 256, 256, 0, stream>>>(Wf[li], wfB[li], n);
  }
  cvt_bf16_pad_kernel<<<(64*64 + 255) / 256, 256, 0, stream>>>(Wt[0], wtB[0], 64, 45, 64);
  for (int li = 1; li < 6; ++li) {
    int n = CFGS[li].cout * CFGS[li].tk * CFGS[li].mid;
    cvt_bf16_kernel<<<(n + 255) / 256, 256, 0, stream>>>(Wt[li], wtB[li], n);
  }
  cvt_bf16_kernel<<<(2097152 + 255) / 256, 256, 0, stream>>>(fc_w, fcwB, 2097152);

  const int BF = 32;                                 // B*S
  const float*          curX = xyzs;
  const unsigned short* curF = nullptr;
  int Lcur = 4, Ncur = 2048;
  bool toA = true;

  for (int li = 0; li < 6; ++li) {
    const Cfg& c = CFGS[li];
    int tr = c.tk / 2;
    int T = 0; for (int t = tr; t < Lcur - tr; t += c.ts) T++;
    int mcur = Ncur / c.ss;
    int p0n = (li < 5) ? CFGS[li+1].p0 : 0;
    int p1n = (li < 5) ? CFGS[li+1].p1 : 0;
    int Lnext = p0n + T + p1n;
    float*          xN = toA ? xyzA : xyzB;
    unsigned short* fN = toA ? featA : featB;
    hipMemsetAsync(xN, 0, (size_t)BF * Lnext * mcur * 3 * sizeof(float), stream);
    hipMemsetAsync(fN, 0, (size_t)BF * Lnext * mcur * c.cout * sizeof(unsigned short), stream);

    int   rows  = BF * mcur;
    int   Ctot  = c.tk * c.mid;
    int   Cpad  = (Ctot + 31) & ~31;                 // only layer 0: 45 -> 64
    float r2    = c.r * c.r;
    int ot = 0;
    for (int t = tr; t < Lcur - tr; t += c.ts, ++ot) {
      int fout = p0n + ot;
      fps_kernel<<<BF, 256, 0, stream>>>(curX, Lcur, t, Ncur, mcur, xN, Lnext, fout);
      for (int s = 0; s < c.tk; ++s) {
        int fr = t - tr + s;
        ballquery_kernel<<<(rows + 255) / 256, 256, 0, stream>>>(
            curX, Lcur, fr, Ncur, xN, Lnext, fout, mcur, r2, idx16, disp16, rows);
        if (c.cin == 0) {
          int tot = rows * c.mid;
          disp_part_kernel<<<(tot + 255) / 256, 256, 0, stream>>>(
              disp16, Wd[li], nf, c.mid, Ctot, s * c.mid, rows);
        } else {
          int ntiles = rows * (c.mid / 32);          // 32 output channels per wave
          feat_gemm_kernel<<<(ntiles + 7) / 8, 256, 0, stream>>>(
              curF, Lcur, fr, Ncur, c.cin, idx16, disp16, wfB[li], Wd[li],
              c.mid, nf, Ctot, s * c.mid, mcur, ntiles);
        }
      }
      if (Cpad != Ctot)   // zero bf16 pad columns so the GEMM K-tail reads zeros
        hipMemsetAsync(nfb, 0, (size_t)rows * Cpad * sizeof(unsigned short), stream);
      bn_relu_kernel<<<Ctot, 256, 0, stream>>>(nf, nfb, rows, Ctot, Cpad);
      int tM = rows / 16;
      int ntiles = tM * (c.cout / 32);               // 1x2 blocked tiles
      gemm_wmma_kernel<<<(ntiles + 7) / 8, 256, 0, stream>>>(
          nfb, Cpad, wtB[li], Cpad, c.cout, nullptr, (li < 5) ? 1 : 0,
          nullptr, fN, mcur, Lnext, fout, ntiles, tM);
    }
    curX = xN; curF = fN; Lcur = Lnext; Ncur = mcur; toA = !toA;
  }

  // Final FC: feats bf16 (32,1,128,2048) @ fc_w^T + fc_b -> d_out f32 (8,4,128,1024)
  {
    int rows = BF * Ncur;            // 32 * 128
    int tM = rows / 16;
    int ntiles = tM * (1024 / 32);
    gemm_wmma_kernel<<<(ntiles + 7) / 8, 256, 0, stream>>>(
        curF, 2048, fcwB, 2048, 1024, fc_b, 0,
        (float*)d_out, nullptr, Ncur, 1, 0, ntiles, tM);
  }
}
